// EarlyExitMamba_58248346468734
// MI455X (gfx1250) — matrix operand order repaired and checked by reading, not compiled, not run on hardware
//
#include <hip/hip_runtime.h>
#include <cstdint>
#include <cstddef>

#define D_MODEL 256
#define D_INNER 512
#define D_STATE 16
#define D_CONV  4
#define DT_RANK 16
#define N_LAYERS 4
#define BATCH 4096
#define SEQ 32

typedef __attribute__((ext_vector_type(16))) __bf16 v16bf;
typedef __attribute__((ext_vector_type(8)))  __bf16 v8bf;
typedef __attribute__((ext_vector_type(8)))  float  v8f;

// ---------- bf16 helpers (manual RNE, no __bf16 scalar arithmetic) ----------
__device__ __forceinline__ unsigned short f2bf(float f) {
  unsigned int u = __float_as_uint(f);
  unsigned int r = u + 0x7FFFu + ((u >> 16) & 1u);
  return (unsigned short)(r >> 16);
}
__device__ __forceinline__ float bf2f(unsigned short h) {
  return __uint_as_float(((unsigned int)h) << 16);
}

// ---------- WMMA operand loaders (CDNA5 ISA 7.12.2 layouts, wave32) ----------
// A 16x32 bf16 from LDS row-major [M][lda]:
//   lanes 0-15: row=lane,    K = kb+0..7  and kb+16..23
//   lanes16-31: row=lane-16, K = kb+8..15 and kb+24..31
__device__ __forceinline__ v16bf ldsA(const unsigned short* sA, int lda,
                                      int mbase, int kt, int lane) {
  int m  = mbase + (lane & 15);
  int kb = kt * 32 + (lane >> 4) * 8;
  const unsigned short* p = sA + m * lda + kb;
  union { v16bf v; v8bf h[2]; } u;
  u.h[0] = *(const v8bf*)(p);        // K = kb .. kb+7
  u.h[1] = *(const v8bf*)(p + 16);   // K = kb+16 .. kb+23
  return u.v;
}
// B 32x16 bf16 from global, weights pre-transposed to [N][ldb(=Kpad)]:
//   lane holds column n = nbase+(lane&15), K = kb..kb+7 and kb+16..kb+23 (kb = kt*32 + (lane>>4)*8)
__device__ __forceinline__ v16bf glbB(const unsigned short* gB, int ldb,
                                      int nbase, int kt, int lane) {
  int n  = nbase + (lane & 15);
  int kb = kt * 32 + (lane >> 4) * 8;
  const unsigned short* p = gB + (size_t)n * ldb + kb;
  union { v16bf v; v8bf h[2]; } u;
  u.h[0] = *(const v8bf*)(p);
  u.h[1] = *(const v8bf*)(p + 16);
  return u.v;
}
__device__ __forceinline__ v8f wmma_bf16(v16bf a, v16bf b, v8f c) {
  return __builtin_amdgcn_wmma_f32_16x16x32_bf16(false, a, false, b,
                                                 (short)0, c, false, false);
}

// ---------- weight convert + transpose: dst[n*Kp+k] = bf16(src[k*N+n]) ----------
__global__ void k_cvt(const float* __restrict__ src, unsigned short* __restrict__ dst,
                      int K, int N, int Kp, int Np) {
  int i = blockIdx.x * 256 + threadIdx.x;
  if (i >= Kp * Np) return;
  int n = i / Kp, k = i - n * Kp;
  float v = (k < K && n < N) ? src[(size_t)k * N + n] : 0.f;
  dst[i] = f2bf(v);
}

// =====================================================================
// Embed + fuse + LayerNorm: one block per sequence b (32 tokens)
// =====================================================================
__global__ __launch_bounds__(256)
void k_embed(const float* __restrict__ x,
             const float* __restrict__ emb_proto, const float* __restrict__ emb_flags,
             const float* __restrict__ emb_dir,
             const float* __restrict__ wlen, const float* __restrict__ blen,
             const float* __restrict__ wiat, const float* __restrict__ biat,
             const unsigned short* __restrict__ wfuse_t,   // [256][160] bf16
             const float* __restrict__ bfuse,
             const float* __restrict__ tok_g, const float* __restrict__ tok_b,
             float* __restrict__ feat) {
  __shared__ unsigned short sCat[SEQ * 160];   // 10 KB
  __shared__ float sOut[SEQ * D_MODEL];        // 32 KB
  const int tid = threadIdx.x, lane = tid & 31, wave = tid >> 5;
  const int b = blockIdx.x;
  const float* xb = x + (size_t)b * SEQ * 5;

  for (int i = tid; i < SEQ * 160; i += 256) {
    int t = i / 160, c = i - t * 160;
    float val = 0.f;
    if (c < 32) {
      int p = (int)xb[t * 5 + 0]; p = p < 0 ? 0 : (p > 255 ? 255 : p);
      val = emb_proto[p * 32 + c];
    } else if (c < 64) {
      val = xb[t * 5 + 1] * wlen[c - 32] + blen[c - 32];
    } else if (c < 96) {
      int f = (int)xb[t * 5 + 2]; f = f < 0 ? 0 : (f > 63 ? 63 : f);
      val = emb_flags[f * 32 + (c - 64)];
    } else if (c < 128) {
      val = xb[t * 5 + 3] * wiat[c - 96] + biat[c - 96];
    } else if (c < 136) {
      int d = (int)xb[t * 5 + 4]; d = d < 0 ? 0 : (d > 1 ? 1 : d);
      val = emb_dir[d * 8 + (c - 128)];
    }
    sCat[i] = f2bf(val);
  }
  __syncthreads();

  // GEMM: [32x160] @ [160x256] -> sOut.  Wave owns N-tiles; one B tile feeds
  // both M-tiles (halves B traffic, back-to-back WMMAs on shared operand).
  for (int nt = wave; nt < 16; nt += 8) {
    v8f c0 = {}, c1 = {};
#pragma unroll
    for (int kt = 0; kt < 5; ++kt) {
      v16bf bm = glbB(wfuse_t, 160, nt * 16, kt, lane);
      c0 = wmma_bf16(ldsA(sCat, 160, 0,  kt, lane), bm, c0);
      c1 = wmma_bf16(ldsA(sCat, 160, 16, kt, lane), bm, c1);
    }
    int n = nt * 16 + (lane & 15);
    int r0 = (lane >> 4) * 8;
    float bias = bfuse[n];
#pragma unroll
    for (int g = 0; g < 8; ++g) {
      sOut[(r0 + g) * D_MODEL + n]      = c0[g] + bias;
      sOut[(r0 + 16 + g) * D_MODEL + n] = c1[g] + bias;
    }
  }
  __syncthreads();

  // LayerNorm over 256 channels, one wave per token (strided)
  float* fo = feat + (size_t)b * SEQ * D_MODEL;
  for (int t = wave; t < SEQ; t += 8) {
    float v[8]; float s = 0.f;
#pragma unroll
    for (int k = 0; k < 8; ++k) { v[k] = sOut[t * D_MODEL + lane + k * 32]; s += v[k]; }
#pragma unroll
    for (int off = 16; off > 0; off >>= 1) s += __shfl_xor(s, off, 32);
    float mean = s * (1.f / 256.f);
    float var = 0.f;
#pragma unroll
    for (int k = 0; k < 8; ++k) { float d = v[k] - mean; var += d * d; }
#pragma unroll
    for (int off = 16; off > 0; off >>= 1) var += __shfl_xor(var, off, 32);
    float inv = rsqrtf(var * (1.f / 256.f) + 1e-5f);
#pragma unroll
    for (int k = 0; k < 8; ++k) {
      int c0i = lane + k * 32;
      fo[t * D_MODEL + c0i] = (v[k] - mean) * inv * tok_g[c0i] + tok_b[c0i];
    }
  }
}

// =====================================================================
// Fused Mamba layer: one block per sequence, everything in LDS (154 KB,
// 2 blocks/WGP). In-place on feat (each block only touches its own rows).
// =====================================================================
__global__ __launch_bounds__(256)
void k_layer(const float* __restrict__ featin, float* __restrict__ featout,
             const unsigned short* __restrict__ inw_t,    // [1024][256]
             const unsigned short* __restrict__ xprj_t,   // [64][512]
             const unsigned short* __restrict__ dtw_t,    // [512][32]
             const unsigned short* __restrict__ outw_t,   // [256][512]
             const float* __restrict__ conv_w, const float* __restrict__ conv_b,
             const float* __restrict__ dt_b,  const float* __restrict__ A_log,
             const float* __restrict__ Dm,
             const float* __restrict__ norm_g, const float* __restrict__ norm_b) {
  extern __shared__ char smem[];
  unsigned short* sA   = (unsigned short*)smem;        // 32*256 feat bf16 (16K)
  unsigned short* sXc  = sA   + SEQ * D_MODEL;         // 32*512 xc bf16 -> ymix bf16 (32K)
  unsigned short* sZ   = sXc  + SEQ * D_INNER;         // 32*512 z bf16 (32K)
  unsigned short* sXcv = sZ   + SEQ * D_INNER;         // 32*512 conv-silu bf16 (32K)
  unsigned short* sDtr = sXcv + SEQ * D_INNER;         // 32*32 dtr bf16 (2K)
  float* sDbl = (float*)(sDtr + SEQ * 32);             // 32*64 dbl f32 (8K)
  float* sRes = sDbl + SEQ * 64;                       // 32*256 f32 (32K), aliases sDt
  unsigned short* sDt = (unsigned short*)sRes;         // 32*512 dt bf16 (32K)

  const int tid = threadIdx.x, lane = tid & 31, wave = tid >> 5;
  const int b = blockIdx.x;
  const float* fin = featin + (size_t)b * SEQ * D_MODEL;
  float* fout = featout + (size_t)b * SEQ * D_MODEL;

  // S1: stage feat as bf16 A-operand
  for (int i = tid; i < SEQ * D_MODEL; i += 256) sA[i] = f2bf(fin[i]);
  __syncthreads();

  // S2: xz = feat @ in_w  (M=32,K=256,N=1024) -> sXc | sZ.
  // Wave owns 8 N-tiles; each B tile is shared by both M-tiles.
  for (int nt = wave; nt < 64; nt += 8) {
    v8f c0 = {}, c1 = {};
#pragma unroll
    for (int kt = 0; kt < 8; ++kt) {
      v16bf bm = glbB(inw_t, D_MODEL, nt * 16, kt, lane);
      c0 = wmma_bf16(ldsA(sA, D_MODEL, 0,  kt, lane), bm, c0);
      c1 = wmma_bf16(ldsA(sA, D_MODEL, 16, kt, lane), bm, c1);
    }
    int n = nt * 16 + (lane & 15);
    int r0 = (lane >> 4) * 8;
    unsigned short* dst = (n < D_INNER) ? (sXc + n) : (sZ + (n - D_INNER));
#pragma unroll
    for (int g = 0; g < 8; ++g) {
      dst[(r0 + g) * D_INNER]      = f2bf(c0[g]);
      dst[(r0 + 16 + g) * D_INNER] = f2bf(c1[g]);
    }
  }
  __syncthreads();

  // S3: causal depthwise conv(4) + SiLU -> sXcv
  for (int i = tid; i < SEQ * D_INNER; i += 256) {
    int t = i >> 9, d = i & (D_INNER - 1);
    float4 cw = *(const float4*)(conv_w + d * 4);
    float acc = conv_b[d];
    float x3 = bf2f(sXc[t * D_INNER + d]);
    float x2 = (t >= 1) ? bf2f(sXc[(t - 1) * D_INNER + d]) : 0.f;
    float x1 = (t >= 2) ? bf2f(sXc[(t - 2) * D_INNER + d]) : 0.f;
    float x0 = (t >= 3) ? bf2f(sXc[(t - 3) * D_INNER + d]) : 0.f;
    acc += x0 * cw.x + x1 * cw.y + x2 * cw.z + x3 * cw.w;
    float s = acc / (1.f + __expf(-acc));   // silu
    sXcv[i] = f2bf(s);
  }
  __syncthreads();

  // S4: dbl = xcv @ xproj (M=32,K=512,N=48 padded 64) -> sDbl f32
  // Tiny weight (traffic negligible) — one tile per wave.
  {
    int mt = wave & 1, nt = wave >> 1;
    v8f c = {};
#pragma unroll
    for (int kt = 0; kt < 16; ++kt)
      c = wmma_bf16(ldsA(sXcv, D_INNER, mt * 16, kt, lane),
                    glbB(xprj_t, D_INNER, nt * 16, kt, lane), c);
    int n = nt * 16 + (lane & 15);
    int r0 = mt * 16 + (lane >> 4) * 8;
#pragma unroll
    for (int g = 0; g < 8; ++g) sDbl[(r0 + g) * 64 + n] = c[g];
  }
  __syncthreads();

  // S4b: stage dtr (K padded 16->32) bf16
  for (int i = tid; i < SEQ * 32; i += 256) {
    int t = i >> 5, k = i & 31;
    sDtr[i] = (k < DT_RANK) ? f2bf(sDbl[t * 64 + k]) : (unsigned short)0;
  }
  __syncthreads();

  // S5: dt = softplus(dtr @ dt_w + dt_b)  (M=32,K=32,N=512) -> sDt bf16
  for (int nt = wave; nt < 32; nt += 8) {
    v16bf bm = glbB(dtw_t, 32, nt * 16, 0, lane);
    v8f c0 = {}, c1 = {};
    c0 = wmma_bf16(ldsA(sDtr, 32, 0,  0, lane), bm, c0);
    c1 = wmma_bf16(ldsA(sDtr, 32, 16, 0, lane), bm, c1);
    int n = nt * 16 + (lane & 15);
    int r0 = (lane >> 4) * 8;
    float bias = dt_b[n];
#pragma unroll
    for (int g = 0; g < 8; ++g) {
      float x0 = c0[g] + bias;
      float x1 = c1[g] + bias;
      sDt[(r0 + g) * D_INNER + n]      = f2bf(x0 > 20.f ? x0 : log1pf(__expf(x0)));
      sDt[(r0 + 16 + g) * D_INNER + n] = f2bf(x1 > 20.f ? x1 : log1pf(__expf(x1)));
    }
  }
  __syncthreads();

  // S6: selective scan (2 channels/thread), fused gating -> ymix bf16 into sXc
  for (int cc = 0; cc < 2; ++cc) {
    int d = tid + cc * 256;
    float Arow[D_STATE];
#pragma unroll
    for (int n = 0; n < D_STATE; ++n) Arow[n] = -__expf(A_log[d * D_STATE + n]);
    float Dmd = Dm[d];
    float h[D_STATE];
#pragma unroll
    for (int n = 0; n < D_STATE; ++n) h[n] = 0.f;
    for (int t = 0; t < SEQ; ++t) {
      float dtv = bf2f(sDt[t * D_INNER + d]);
      float xv  = bf2f(sXcv[t * D_INNER + d]);
      float zv  = bf2f(sZ[t * D_INNER + d]);
      float dx = dtv * xv;
      float acc = 0.f;
#pragma unroll
      for (int n = 0; n < D_STATE; ++n) {
        float Bn = sDbl[t * 64 + DT_RANK + n];
        float Cn = sDbl[t * 64 + DT_RANK + D_STATE + n];
        h[n] = h[n] * __expf(dtv * Arow[n]) + dx * Bn;
        acc += h[n] * Cn;
      }
      float y = acc + xv * Dmd;
      float ymix = y * (zv / (1.f + __expf(-zv)));
      sXc[t * D_INNER + d] = f2bf(ymix);     // sXc reused as out-proj A operand
    }
  }
  __syncthreads();

  // S7: out = ymix @ out_w + residual (M=32,K=512,N=256) -> sRes f32
  for (int nt = wave; nt < 16; nt += 8) {
    v8f c0 = {}, c1 = {};
#pragma unroll
    for (int kt = 0; kt < 16; ++kt) {
      v16bf bm = glbB(outw_t, D_INNER, nt * 16, kt, lane);
      c0 = wmma_bf16(ldsA(sXc, D_INNER, 0,  kt, lane), bm, c0);
      c1 = wmma_bf16(ldsA(sXc, D_INNER, 16, kt, lane), bm, c1);
    }
    int n = nt * 16 + (lane & 15);
    int r0 = (lane >> 4) * 8;
#pragma unroll
    for (int g = 0; g < 8; ++g) {
      int ra = r0 + g, rb = r0 + 16 + g;
      sRes[ra * D_MODEL + n] = c0[g] + fin[ra * D_MODEL + n];
      sRes[rb * D_MODEL + n] = c1[g] + fin[rb * D_MODEL + n];
    }
  }
  __syncthreads();

  // S8: LayerNorm + writeback (in place is safe: block-private rows)
  for (int t = wave; t < SEQ; t += 8) {
    float v[8]; float s = 0.f;
#pragma unroll
    for (int k = 0; k < 8; ++k) { v[k] = sRes[t * D_MODEL + lane + k * 32]; s += v[k]; }
#pragma unroll
    for (int off = 16; off > 0; off >>= 1) s += __shfl_xor(s, off, 32);
    float mean = s * (1.f / 256.f);
    float var = 0.f;
#pragma unroll
    for (int k = 0; k < 8; ++k) { float d = v[k] - mean; var += d * d; }
#pragma unroll
    for (int off = 16; off > 0; off >>= 1) var += __shfl_xor(var, off, 32);
    float inv = rsqrtf(var * (1.f / 256.f) + 1e-5f);
#pragma unroll
    for (int k = 0; k < 8; ++k) {
      int c0i = lane + k * 32;
      fout[t * D_MODEL + c0i] = (v[k] - mean) * inv * norm_g[c0i] + norm_b[c0i];
    }
  }
}

// =====================================================================
// Heads: means over first 8/16/32 tokens, 3 MLPs 256->64(relu)->2
// =====================================================================
__global__ __launch_bounds__(256)
void k_heads(const float* __restrict__ feat,
             const float* __restrict__ cls_w1, const float* __restrict__ cls_b1,
             const float* __restrict__ cls_w2, const float* __restrict__ cls_b2,
             float* __restrict__ out) {
  __shared__ float sF[SEQ * D_MODEL];
  __shared__ float sP[3 * D_MODEL];
  __shared__ float sH[3 * 64];
  const int tid = threadIdx.x;
  const int b = blockIdx.x;
  const float* fb = feat + (size_t)b * SEQ * D_MODEL;
  for (int i = tid; i < SEQ * D_MODEL; i += 256) sF[i] = fb[i];
  __syncthreads();
  {
    float s8 = 0.f, s16 = 0.f, s32 = 0.f;
    for (int t = 0; t < SEQ; ++t) {
      float v = sF[t * D_MODEL + tid];
      s32 += v; if (t < 16) s16 += v; if (t < 8) s8 += v;
    }
    sP[0 * D_MODEL + tid] = s8 * (1.f / 8.f);
    sP[1 * D_MODEL + tid] = s16 * (1.f / 16.f);
    sP[2 * D_MODEL + tid] = s32 * (1.f / 32.f);
  }
  __syncthreads();
  if (tid < 192) {
    int head = tid >> 6, j = tid & 63;
    float a = cls_b1[head * 64 + j];
    const float* w = cls_w1 + (size_t)head * D_MODEL * 64 + j;
    const float* p = sP + head * D_MODEL;
    for (int c = 0; c < D_MODEL; ++c) a += p[c] * w[(size_t)c * 64];
    sH[head * 64 + j] = a > 0.f ? a : 0.f;
  }
  __syncthreads();
  if (tid < 6) {
    int head = tid >> 1, o = tid & 1;
    float a = cls_b2[head * 2 + o];
    const float* w = cls_w2 + (size_t)head * 64 * 2 + o;
    const float* h = sH + head * 64;
    for (int j = 0; j < 64; ++j) a += h[j] * w[(size_t)j * 2];
    out[(size_t)head * BATCH * 2 + (size_t)b * 2 + o] = a;
  }
}

// =====================================================================
extern "C" void kernel_launch(void* const* d_in, const int* in_sizes, int n_in,
                              void* d_out, int out_size, void* d_ws, size_t ws_size,
                              hipStream_t stream) {
  const float* x         = (const float*)d_in[0];
  const float* emb_proto = (const float*)d_in[1];
  const float* emb_flags = (const float*)d_in[2];
  const float* emb_dir   = (const float*)d_in[3];
  const float* wlen      = (const float*)d_in[4];
  const float* blen      = (const float*)d_in[5];
  const float* wiat      = (const float*)d_in[6];
  const float* biat      = (const float*)d_in[7];
  const float* wfuse     = (const float*)d_in[8];
  const float* bfuse     = (const float*)d_in[9];
  const float* tok_g     = (const float*)d_in[10];
  const float* tok_b     = (const float*)d_in[11];
  const float* norm_g    = (const float*)d_in[12];
  const float* norm_b    = (const float*)d_in[13];
  const float* in_w      = (const float*)d_in[14];
  const float* conv_w    = (const float*)d_in[15];
  const float* conv_b    = (const float*)d_in[16];
  const float* xproj_w   = (const float*)d_in[17];
  const float* dt_w      = (const float*)d_in[18];
  const float* dt_b      = (const float*)d_in[19];
  const float* A_log     = (const float*)d_in[20];
  const float* Dm        = (const float*)d_in[21];
  const float* out_w     = (const float*)d_in[22];
  const float* cls_w1    = (const float*)d_in[23];
  const float* cls_b1    = (const float*)d_in[24];
  const float* cls_w2    = (const float*)d_in[25];
  const float* cls_b2    = (const float*)d_in[26];
  float* out = (float*)d_out;

  char* ws = (char*)d_ws;
  size_t off = 0;
  auto carve = [&](size_t bytes) -> void* {
    void* p = ws + off;
    off += (bytes + 255) & ~(size_t)255;
    return p;
  };
  float* feat = (float*)carve((size_t)BATCH * SEQ * D_MODEL * sizeof(float));
  unsigned short* wfuse_t = (unsigned short*)carve((size_t)256 * 160 * 2);
  unsigned short *inw_t[N_LAYERS], *xprj_t[N_LAYERS], *dtw_t[N_LAYERS], *outw_t[N_LAYERS];
  for (int l = 0; l < N_LAYERS; ++l) {
    inw_t[l]  = (unsigned short*)carve((size_t)1024 * 256 * 2);
    xprj_t[l] = (unsigned short*)carve((size_t)64 * 512 * 2);
    dtw_t[l]  = (unsigned short*)carve((size_t)512 * 32 * 2);
    outw_t[l] = (unsigned short*)carve((size_t)256 * 512 * 2);
  }

  auto cvt = [&](const float* src, unsigned short* dst, int K, int N, int Kp, int Np) {
    int total = Kp * Np;
    k_cvt<<<(total + 255) / 256, 256, 0, stream>>>(src, dst, K, N, Kp, Np);
  };
  cvt(wfuse, wfuse_t, 136, 256, 160, 256);
  for (int l = 0; l < N_LAYERS; ++l) {
    cvt(in_w    + (size_t)l * 256 * 1024, inw_t[l],  256, 1024, 256, 1024);
    cvt(xproj_w + (size_t)l * 512 * 48,   xprj_t[l], 512, 48,   512, 64);
    cvt(dt_w    + (size_t)l * 16 * 512,   dtw_t[l],  16,  512,  32,  512);
    cvt(out_w   + (size_t)l * 512 * 256,  outw_t[l], 512, 256,  512, 256);
  }

  k_embed<<<BATCH, 256, 0, stream>>>(x, emb_proto, emb_flags, emb_dir,
                                     wlen, blen, wiat, biat,
                                     wfuse_t, bfuse, tok_g, tok_b, feat);

  const size_t layer_lds =
      (size_t)(SEQ * D_MODEL + 3 * SEQ * D_INNER + SEQ * 32) * sizeof(unsigned short)
      + (size_t)(SEQ * 64) * sizeof(float)
      + (size_t)(SEQ * D_INNER) * sizeof(unsigned short);  // = 157,696 bytes

  for (int l = 0; l < N_LAYERS; ++l) {
    k_layer<<<BATCH, 256, layer_lds, stream>>>(
        feat, feat, inw_t[l], xprj_t[l], dtw_t[l], outw_t[l],
        conv_w + (size_t)l * 512 * 4, conv_b + (size_t)l * 512,
        dt_b + (size_t)l * 512, A_log + (size_t)l * 512 * 16,
        Dm + (size_t)l * 512, norm_g, norm_b);
  }

  k_heads<<<BATCH, 256, 0, stream>>>(feat, cls_w1, cls_b1, cls_w2, cls_b2, out);
}
